// MultiHeadAttention_34548716929133
// MI455X (gfx1250) — matrix operand order repaired and checked by reading, not compiled, not run on hardware
//
#include <hip/hip_runtime.h>

#define DMODEL 1024
#define NHEAD  16
#define DK     64
#define BATCH  2
#define SEQ    2048
#define MTOT   (BATCH * SEQ)   // 4096 rows

typedef __bf16 bf16;
typedef __attribute__((ext_vector_type(16))) __bf16 v16bf;
typedef __attribute__((ext_vector_type(8)))  __bf16 v8bf;
typedef __attribute__((ext_vector_type(4)))  __bf16 v4bf;
typedef __attribute__((ext_vector_type(8)))  float  v8f;

static __device__ __forceinline__ v16bf cat8(v8bf lo, v8bf hi) {
  return __builtin_shufflevector(lo, hi, 0,1,2,3,4,5,6,7,8,9,10,11,12,13,14,15);
}
static __device__ __forceinline__ v8f wmma_bf16(v16bf a, v16bf b, v8f c) {
  // D = A(16x32 bf16) * B(32x16 bf16) + C(16x16 f32)
  return __builtin_amdgcn_wmma_f32_16x16x32_bf16(false, a, false, b, (short)0, c,
                                                 false, false);
}

// ---------------------------------------------------------------------------
// f32 -> bf16 conversion (vectorized x4)
// ---------------------------------------------------------------------------
__global__ void __launch_bounds__(256)
cvt_f32_bf16(const float* __restrict__ s, bf16* __restrict__ d, int n4) {
  int i = blockIdx.x * 256 + threadIdx.x;
  if (i >= n4) return;
  float4 f = reinterpret_cast<const float4*>(s)[i];
  v4bf o;
  o[0] = (bf16)f.x; o[1] = (bf16)f.y; o[2] = (bf16)f.z; o[3] = (bf16)f.w;
  reinterpret_cast<v4bf*>(d)[i] = o;
}

// ---------------------------------------------------------------------------
// GEMM: C[M,N] = A[M,K] * W[N,K]^T + bias[N]   (M=4096, N=K=1024, bf16 in)
// One wave per 32x32 output tile: 2 A-frags x 2 B-frags -> 4 WMMAs per K=32
// step; halves per-WMMA load bytes vs 16x16 tiling.
// MODE 0: store bf16 [M,N]
// MODE 1: store bf16 [M,N] scaled by 1/sqrt(DK) (Q projection)
// MODE 2: store bf16 transposed per head -> [B, H, DK, S] (V projection)
// MODE 3: store f32 [M,N] (final output projection)
// ---------------------------------------------------------------------------
template <int MODE>
__global__ void __launch_bounds__(128)
gemm_bias(const bf16* __restrict__ A, const bf16* __restrict__ W,
          const float* __restrict__ bias, void* __restrict__ out) {
  const int lane = threadIdx.x & 31;
  const int wv   = threadIdx.x >> 5;
  const int gid  = blockIdx.x * 4 + wv;   // 4096 tiles of 32x32
  const int mt   = gid >> 5;              // 128 row tiles
  const int nt   = gid & 31;              // 32  col tiles
  const int half = lane >> 4;
  const int lm   = lane & 15;

  const bf16* ap0 = A + (size_t)(mt * 32 + lm) * DMODEL + half * 16;
  const bf16* ap1 = ap0 + (size_t)16 * DMODEL;
  const bf16* wp0 = W + (size_t)(nt * 32 + lm) * DMODEL + half * 16;
  const bf16* wp1 = wp0 + (size_t)16 * DMODEL;

  v8f acc00 = {}, acc01 = {}, acc10 = {}, acc11 = {};
#pragma unroll 2
  for (int k0 = 0; k0 < DMODEL; k0 += 32) {
    __builtin_prefetch(ap0 + k0 + 256, 0, 1);
    __builtin_prefetch(wp0 + k0 + 256, 0, 1);
    v16bf a0 = cat8(*reinterpret_cast<const v8bf*>(ap0 + k0),
                    *reinterpret_cast<const v8bf*>(ap0 + k0 + 8));
    v16bf a1 = cat8(*reinterpret_cast<const v8bf*>(ap1 + k0),
                    *reinterpret_cast<const v8bf*>(ap1 + k0 + 8));
    v16bf b0 = *reinterpret_cast<const v16bf*>(wp0 + k0);
    v16bf b1 = *reinterpret_cast<const v16bf*>(wp1 + k0);
    acc00 = wmma_bf16(a0, b0, acc00);
    acc01 = wmma_bf16(a0, b1, acc01);
    acc10 = wmma_bf16(a1, b0, acc10);
    acc11 = wmma_bf16(a1, b1, acc11);
  }

  v8f* accs[4] = { &acc00, &acc01, &acc10, &acc11 };
#pragma unroll
  for (int i = 0; i < 2; ++i) {
#pragma unroll
    for (int j = 0; j < 2; ++j) {
      const v8f& acc = *accs[i * 2 + j];
      const int col  = nt * 32 + j * 16 + lm;
      const float bv = bias[col];
#pragma unroll
      for (int r = 0; r < 8; ++r) {
        const int row = mt * 32 + i * 16 + r + 8 * half;
        float v = acc[r] + bv;
        if (MODE == 0) {
          reinterpret_cast<bf16*>(out)[(size_t)row * DMODEL + col] = (bf16)v;
        } else if (MODE == 1) {
          reinterpret_cast<bf16*>(out)[(size_t)row * DMODEL + col] =
              (bf16)(v * 0.125f);
        } else if (MODE == 2) {
          const int b = row >> 11, s = row & (SEQ - 1);
          const int h = col >> 6,  d = col & (DK - 1);
          reinterpret_cast<bf16*>(out)
              [((size_t)((b * NHEAD + h) * DK + d)) * SEQ + s] = (bf16)v;
        } else {
          reinterpret_cast<float*>(out)[(size_t)row * DMODEL + col] = v;
        }
      }
    }
  }
}

// ---------------------------------------------------------------------------
// Flash attention: one wave per (b, h, 16-query tile). Q pre-scaled by 1/8.
// Per 32-key step: 4 score WMMAs, online softmax, P staged via LDS into
// A-operand layout, 4 PV WMMAs. V fragments are double-buffered in LDS via
// GLOBAL_LOAD_ASYNC_TO_LDS_B128 (ASYNCcnt), overlapping the next key block's
// V traffic with score/softmax work.
// ---------------------------------------------------------------------------
__global__ void __launch_bounds__(32)
attn_fwd(const bf16* __restrict__ Qp, const bf16* __restrict__ Kp,
         const bf16* __restrict__ Vt, bf16* __restrict__ ctx) {
  __shared__ __align__(16) bf16 Plds[16][40];        // 16 q-rows x 32 keys
  __shared__ __align__(16) bf16 vbuf[2][4][32][16];  // dbl-buf V frags, 8 KB

  const int lane = threadIdx.x;
  const int half = lane >> 4;
  const int lm   = lane & 15;
  const int id   = blockIdx.x;             // 4096 = B*H*(S/16)
  const int qt   = id & (SEQ / 16 - 1);
  const int h    = (id >> 7) & (NHEAD - 1);
  const int b    = id >> 11;

  // Q A-fragments for d-slices [0,32) and [32,64)
  const bf16* qbase = Qp + (size_t)(b * SEQ + qt * 16 + lm) * DMODEL + h * DK;
  v16bf aq[2];
#pragma unroll
  for (int dsl = 0; dsl < 2; ++dsl) {
    v8bf lo = *reinterpret_cast<const v8bf*>(qbase + dsl * 32 + half * 16);
    v8bf hi = *reinterpret_cast<const v8bf*>(qbase + dsl * 32 + half * 16 + 8);
    aq[dsl] = cat8(lo, hi);
  }

  v8f o[4];
  float rmax[8], rsum[8];
#pragma unroll
  for (int nt = 0; nt < 4; ++nt) o[nt] = v8f{};
#pragma unroll
  for (int r = 0; r < 8; ++r) { rmax[r] = -3.0e38f; rsum[r] = 0.f; }

  const bf16* kbase = Kp + (size_t)(b * SEQ) * DMODEL + h * DK;
  const bf16* vbase = Vt + ((size_t)(b * NHEAD + h) * DK) * SEQ;

  // Issue async V-fragment loads (8 x 16B per lane) for key block kb2 -> bufi.
  auto issue_v = [&](int bufi, int kb2) {
#pragma unroll
    for (int nt = 0; nt < 4; ++nt) {
#pragma unroll
      for (int c = 0; c < 2; ++c) {
        const bf16* gp =
            vbase + (size_t)(nt * 16 + lm) * SEQ + kb2 + half * 16 + c * 8;
        unsigned ldsa = (unsigned)(uintptr_t)&vbuf[bufi][nt][lane][c * 8];
        asm volatile("global_load_async_to_lds_b128 %0, %1, off"
                     :: "v"(ldsa), "v"((unsigned long long)(uintptr_t)gp)
                     : "memory");
      }
    }
  };

  int buf = 0;
  issue_v(0, 0);   // prologue: key block 0 -> buffer 0

  for (int kb = 0; kb < SEQ; kb += 32) {
    const bool more = (kb + 32) < SEQ;
    if (more) issue_v(buf ^ 1, kb + 32);   // overlap next block's V traffic

    // ---- scores for two 16-key tiles (Q pre-scaled: no extra scaling) ----
    v8f s[2];
#pragma unroll
    for (int t = 0; t < 2; ++t) {
      const bf16* kp = kbase + (size_t)(kb + t * 16 + lm) * DMODEL + half * 16;
      v16bf b0 = *reinterpret_cast<const v16bf*>(kp);        // d 0..31 slice
      v16bf b1 = *reinterpret_cast<const v16bf*>(kp + 32);   // d 32..63 slice
      v8f acc = {};
      acc  = wmma_bf16(aq[0], b0, acc);
      s[t] = wmma_bf16(aq[1], b1, acc);
    }

    // ---- online softmax (row = r + 8*half lives across 16-lane group) ----
#pragma unroll
    for (int r = 0; r < 8; ++r) {
      float v0 = s[0][r], v1 = s[1][r];
      float bm = fmaxf(v0, v1);
#pragma unroll
      for (int m = 1; m < 16; m <<= 1) bm = fmaxf(bm, __shfl_xor(bm, m, 32));
      float mnew = fmaxf(rmax[r], bm);
      float corr = __expf(rmax[r] - mnew);
      float p0 = __expf(v0 - mnew);
      float p1 = __expf(v1 - mnew);
      float ps = p0 + p1;
#pragma unroll
      for (int m = 1; m < 16; m <<= 1) ps += __shfl_xor(ps, m, 32);
      rsum[r] = rsum[r] * corr + ps;
      rmax[r] = mnew;
#pragma unroll
      for (int nt = 0; nt < 4; ++nt) o[nt][r] *= corr;
      Plds[r + 8 * half][lm]      = (bf16)p0;
      Plds[r + 8 * half][16 + lm] = (bf16)p1;
    }
    asm volatile("s_wait_dscnt 0" ::: "memory");

    // ---- reload P in A-operand layout (row lm, K slices per lane half) ----
    v8bf plo = *reinterpret_cast<const v8bf*>(&Plds[lm][half * 16]);
    v8bf phi = *reinterpret_cast<const v8bf*>(&Plds[lm][half * 16 + 8]);
    v16bf pa = cat8(plo, phi);

    // ---- wait for current V buffer (async loads complete in order: the 8
    //      older requests are done once <= 8 newer ones remain outstanding) --
    if (more) asm volatile("s_wait_asynccnt 0x8" ::: "memory");
    else      asm volatile("s_wait_asynccnt 0x0" ::: "memory");

    // ---- O += P(16x32) * V(32x64) from LDS-staged fragments ----
#pragma unroll
    for (int nt = 0; nt < 4; ++nt) {
      v16bf vb = *reinterpret_cast<const v16bf*>(&vbuf[buf][nt][lane][0]);
      o[nt] = wmma_bf16(pa, vb, o[nt]);
    }
    buf ^= 1;
  }

  // ---- normalize and store context bf16 [B,S,D] ----
#pragma unroll
  for (int r = 0; r < 8; ++r) {
    float inv = 1.0f / rsum[r];
    const size_t row = (size_t)(b * SEQ + qt * 16 + r + 8 * half);
#pragma unroll
    for (int nt = 0; nt < 4; ++nt)
      ctx[row * DMODEL + h * DK + nt * 16 + lm] = (bf16)(o[nt][r] * inv);
  }
}

// ---------------------------------------------------------------------------
// Host launch
// ---------------------------------------------------------------------------
extern "C" void kernel_launch(void* const* d_in, const int* in_sizes, int n_in,
                              void* d_out, int out_size, void* d_ws, size_t ws_size,
                              hipStream_t stream) {
  (void)in_sizes; (void)n_in; (void)out_size; (void)ws_size;
  // inputs: x, q, k, v, mask, w_q, b_q, w_k, b_k, w_v, b_v, w_o, b_o
  const float* q  = (const float*)d_in[1];
  const float* k  = (const float*)d_in[2];
  const float* v  = (const float*)d_in[3];
  const float* wq = (const float*)d_in[5];
  const float* bq = (const float*)d_in[6];
  const float* wk = (const float*)d_in[7];
  const float* bk = (const float*)d_in[8];
  const float* wo = (const float*)d_in[11];
  const float* bo = (const float*)d_in[12];
  // x, mask, w_v, b_v intentionally unused (faithful to reference bugs).

  char* ws = (char*)d_ws;
  const size_t MB = (size_t)1 << 20;
  bf16* qb  = (bf16*)(ws +  0 * MB);   // q  bf16   [4096,1024]  8 MB
  bf16* kbf = (bf16*)(ws +  8 * MB);   // k  bf16                8 MB
  bf16* vbf = (bf16*)(ws + 16 * MB);   // v  bf16                8 MB
  bf16* Qp  = (bf16*)(ws + 24 * MB);   // Q proj (scaled)        8 MB
  bf16* Kp  = (bf16*)(ws + 32 * MB);   // K proj                 8 MB
  bf16* Vt  = (bf16*)(ws + 40 * MB);   // V proj transposed      8 MB
  bf16* ctx = (bf16*)(ws + 48 * MB);   // attention context      8 MB
  bf16* wqb = (bf16*)(ws + 56 * MB);   // w_q bf16               2 MB
  bf16* wkb = (bf16*)(ws + 58 * MB);   // w_k bf16               2 MB
  bf16* wob = (bf16*)(ws + 60 * MB);   // w_o bf16               2 MB

  const int nAct = MTOT * DMODEL;      // 4194304
  const int nWgt = DMODEL * DMODEL;    // 1048576
  cvt_f32_bf16<<<dim3(nAct / 4 / 256), dim3(256), 0, stream>>>(q,  qb,  nAct / 4);
  cvt_f32_bf16<<<dim3(nAct / 4 / 256), dim3(256), 0, stream>>>(k,  kbf, nAct / 4);
  cvt_f32_bf16<<<dim3(nAct / 4 / 256), dim3(256), 0, stream>>>(v,  vbf, nAct / 4);
  cvt_f32_bf16<<<dim3(nWgt / 4 / 256), dim3(256), 0, stream>>>(wq, wqb, nWgt / 4);
  cvt_f32_bf16<<<dim3(nWgt / 4 / 256), dim3(256), 0, stream>>>(wk, wkb, nWgt / 4);
  cvt_f32_bf16<<<dim3(nWgt / 4 / 256), dim3(256), 0, stream>>>(wo, wob, nWgt / 4);

  const dim3 ggrid(1024), gblk(128);   // 4096 waves, one 32x32 tile each
  gemm_bias<1><<<ggrid, gblk, 0, stream>>>(qb,  wqb, bq, (void*)Qp);  // Q, pre-scaled
  gemm_bias<0><<<ggrid, gblk, 0, stream>>>(kbf, wkb, bk, (void*)Kp);  // K
  gemm_bias<2><<<ggrid, gblk, 0, stream>>>(vbf, wkb, bk, (void*)Vt);  // V uses w_k (ref bug)

  attn_fwd<<<dim3(4096), dim3(32), 0, stream>>>(Qp, Kp, Vt, ctx);

  gemm_bias<3><<<ggrid, gblk, 0, stream>>>(ctx, wob, bo, d_out);      // f32 out
}
